// SinkhornKLLoss_49615462203961
// MI455X (gfx1250) — compile-verified, matrix operand order checked by reading, not run on hardware
//
#include <hip/hip_runtime.h>
#include <hip/hip_bf16.h>

typedef __attribute__((ext_vector_type(16))) _Float16 v16h;
typedef __attribute__((ext_vector_type(8)))  _Float16 v8h;
typedef __attribute__((ext_vector_type(8)))  float    v8f;
typedef __attribute__((ext_vector_type(4)))  unsigned int u32x4;
typedef __attribute__((ext_vector_type(4)))  int i32x4;
typedef __attribute__((ext_vector_type(8)))  int i32x8;

#define N_S   2048
#define D_F   256
#define REG_  0.1f
#define INV_REG 10.0f
#define EPS_  1e-10f
#define PROX  5
#define SINK  100
#define PQ    (1.0f/2048.0f)       /* uniform marginals p = q = 1/n */

// ---------------- reductions (wave32) ----------------
__device__ __forceinline__ float warpSum(float v) {
#pragma unroll
  for (int o = 16; o > 0; o >>= 1) v += __shfl_xor(v, o, 32);
  return v;
}
__device__ __forceinline__ float warpMax(float v) {
#pragma unroll
  for (int o = 16; o > 0; o >>= 1) v = fmaxf(v, __shfl_xor(v, o, 32));
  return v;
}
__device__ __forceinline__ float blockSum256(float v, float* red) {
  v = warpSum(v);
  int lane = threadIdx.x & 31, w = threadIdx.x >> 5;
  __syncthreads();
  if (lane == 0) red[w] = v;
  __syncthreads();
  float r = 0.f;
#pragma unroll
  for (int i = 0; i < 8; ++i) r += red[i];
  return r;
}

// ---------------- setup kernels ----------------
__global__ void k_cvt(const float* __restrict__ s, _Float16* __restrict__ d, int n) {
  int i = blockIdx.x * blockDim.x + threadIdx.x;
  if (i < n) d[i] = (_Float16)s[i];
}

__global__ __launch_bounds__(256) void k_rowsq(const float* __restrict__ X,
                                               float* __restrict__ sq) {
  int wave = (blockIdx.x * 256 + threadIdx.x) >> 5;   // one wave per row
  int lane = threadIdx.x & 31;
  const float* row = X + (size_t)wave * D_F;
  float s = 0.f;
  for (int k = lane; k < D_F; k += 32) { float t = row[k]; s += t * t; }
  s = warpSum(s);
  if (lane == 0) sq[wave] = s;
}

__global__ void k_fill(float* __restrict__ p, float val, int n) {
  int i = blockIdx.x * blockDim.x + threadIdx.x;
  if (i < n) p[i] = val;
}

// ---------------- Tensor Data Mover tile load ----------------
// 2D tile: tile_dim0 = 32 halfs (64B contiguous), tile_dim1 = 128 rows,
// row stride = 256 halfs. LDS pad: 4 DWORDs (16B) after every 16 DWORDs (64B)
// -> LDS row stride 80B, matching the 40-half padded tile layout.
__device__ __forceinline__ void tdm_tile_load(const _Float16* gsrc, unsigned lds_addr) {
  unsigned long long ga = (unsigned long long)(const void*)gsrc;
  u32x4 g0;
  g0.x = 1u;                                            // count=1 (valid user D#)
  g0.y = lds_addr;                                      // LDS byte address
  g0.z = (unsigned)ga;                                  // global_addr[31:0]
  g0.w = (unsigned)((ga >> 32) & 0x01FFFFFFu)           // global_addr[56:32]
       | 0x80000000u;                                   // type = 2 ("image")

  const unsigned td0 = 1u << 24, td1 = 1u << 24;        // huge tensor dims: no OOB clip
  const unsigned tile0 = 32u, tile1 = 128u, tile2 = 0u;
  const unsigned long long s0 = (unsigned long long)D_F; // dim0 stride (elements)
  i32x8 g1;
  g1[0] = (int)((1u << 16)      // data_size = 1 -> 2 bytes
              | (1u << 20)      // pad_enable
              | (3u << 22)      // pad_interval: 16 DWORDs
              | (3u << 25));    // pad_amount:   4 DWORDs
  g1[1] = (int)((td0 & 0xFFFFu) << 16);                    // tensor_dim0[15:0]
  g1[2] = (int)((td0 >> 16) | ((td1 & 0xFFFFu) << 16));    // td0[31:16] | td1[15:0]
  g1[3] = (int)((td1 >> 16) | (tile0 << 16));              // td1[31:16] | tile_dim0
  g1[4] = (int)(tile1 | (tile2 << 16));                    // tile_dim1 | tile_dim2
  g1[5] = (int)(s0 & 0xFFFFFFFFull);                       // stride0[31:0]
  g1[6] = (int)((s0 >> 32) & 0xFFFFull);                   // stride0[47:32] | stride1[15:0]
  g1[7] = 0;                                               // stride1[47:16]
  i32x4 g2; g2[0] = (int)(1u << 24); g2[1] = 0; g2[2] = 0; g2[3] = 0; // tensor_dim2 big
  i32x4 g3; g3[0] = 0; g3[1] = 0; g3[2] = 0; g3[3] = 0;
#if defined(__clang_major__) && (__clang_major__ >= 23)
  i32x8 g4 = {0, 0, 0, 0, 0, 0, 0, 0};
  __builtin_amdgcn_tensor_load_to_lds(g0, g1, g2, g3, g4, 0);
#else
  __builtin_amdgcn_tensor_load_to_lds(g0, g1, g2, g3, 0);
#endif
}

// ---------------- WMMA GEMM: G = -2 * A * B^T  (A,B: [2048 x 256] f16) ----------------
// Block tile 128x128, BK=32, 8 waves (4x2), each wave: 2x4 = 8 wmma 16x16 tiles.
// Tiles staged by the Tensor Data Mover (wave 0 issues), double-buffered in LDS.
__global__ __launch_bounds__(256) void k_gemm(const _Float16* __restrict__ A,
                                              const _Float16* __restrict__ B,
                                              float* __restrict__ G) {
  __shared__ _Float16 As[2 * 128 * 40];   // 2 x 10240B, padded rows (80B stride)
  __shared__ _Float16 Bs[2 * 128 * 40];
  const int t    = threadIdx.x;
  const int lane = t & 31;
  const int wid  = t >> 5;
  const int wm   = wid & 3;          // wave row group (4)
  const int wn   = wid >> 2;         // wave col group (2)
  const int m0   = blockIdx.y * 128;
  const int n0   = blockIdx.x * 128;
  const int hi   = lane >> 4;        // lane half (K-group select per ISA layout)
  const int l15  = lane & 15;

  const unsigned ldsA = (unsigned)(unsigned long long)(void*)As;
  const unsigned ldsB = (unsigned)(unsigned long long)(void*)Bs;
  const unsigned bufB = 128u * 40u * 2u;   // bytes per buffer

  v8f acc[2][4] = {};

  if (wid == 0) {                      // prologue: DMA pair 0 into buffer 0
    tdm_tile_load(A + m0 * D_F, ldsA);
    tdm_tile_load(B + n0 * D_F, ldsB);
  }

  int cur = 0;
  for (int kt = 0; kt < D_F / 32; ++kt) {
    if (wid == 0) {
      if (kt + 1 < D_F / 32) {         // stream next pair into the other buffer
        int koff = (kt + 1) * 32;
        unsigned nb = (unsigned)(cur ^ 1) * bufB;
        tdm_tile_load(A + m0 * D_F + koff, ldsA + nb);
        tdm_tile_load(B + n0 * D_F + koff, ldsB + nb);
        __builtin_amdgcn_s_wait_tensorcnt(2);   // current pair complete
      } else {
        __builtin_amdgcn_s_wait_tensorcnt(0);   // last pair complete
      }
    }
    __syncthreads();                   // current buffers visible to all waves

    const _Float16* Ab = As + cur * (128 * 40);
    const _Float16* Bb = Bs + cur * (128 * 40);

    v16h af[2], bf[4];
#pragma unroll
    for (int f = 0; f < 2; ++f) {
      int r = wm * 32 + f * 16 + l15;
      v8h lo = *(const v8h*)&Ab[r * 40 + hi * 8];        // K = hi*8 .. +7
      v8h h2 = *(const v8h*)&Ab[r * 40 + 16 + hi * 8];   // K = 16+hi*8 .. +7
      af[f] = __builtin_shufflevector(lo, h2, 0,1,2,3,4,5,6,7,8,9,10,11,12,13,14,15);
    }
#pragma unroll
    for (int g = 0; g < 4; ++g) {
      int c = wn * 64 + g * 16 + l15;
      v8h lo = *(const v8h*)&Bb[c * 40 + hi * 8];
      v8h h2 = *(const v8h*)&Bb[c * 40 + 16 + hi * 8];
      bf[g] = __builtin_shufflevector(lo, h2, 0,1,2,3,4,5,6,7,8,9,10,11,12,13,14,15);
    }
#pragma unroll
    for (int f = 0; f < 2; ++f)
#pragma unroll
      for (int g = 0; g < 4; ++g)
        acc[f][g] = __builtin_amdgcn_wmma_f32_16x16x32_f16(
            false, af[f], false, bf[g], (short)0, acc[f][g], false, false);

    __syncthreads();                   // all waves done with `cur` before refill
    cur ^= 1;
  }

  // epilogue: C/D layout -> row = base + v + 8*hi, col = base + (lane&15)
#pragma unroll
  for (int f = 0; f < 2; ++f) {
    int row0 = m0 + wm * 32 + f * 16 + hi * 8;
#pragma unroll
    for (int g = 0; g < 4; ++g) {
      int col = n0 + wn * 64 + g * 16 + l15;
#pragma unroll
      for (int v = 0; v < 8; ++v)
        G[(row0 + v) * N_S + col] = -2.0f * acc[f][g][v];
    }
  }
}

// ---------------- M = sqa[i]+sqb[j]+G - reg*log(P+eps); block-partial max ----------------
__global__ __launch_bounds__(256) void k_partmax(const float* __restrict__ G,
                                                 const float* __restrict__ P,
                                                 const float* __restrict__ sqa,
                                                 const float* __restrict__ sqb,
                                                 float* __restrict__ bmax) {
  __shared__ float red[8];
  int base = blockIdx.x * 2048;
  float m = -3.4e38f;
#pragma unroll
  for (int e = 0; e < 8; ++e) {
    int idx = base + threadIdx.x + e * 256;
    int i = idx >> 11, j = idx & (N_S - 1);
    float Mv = sqa[i] + sqb[j] + G[idx] - REG_ * logf(P[idx] + EPS_);
    m = fmaxf(m, Mv);
  }
  m = warpMax(m);
  int lane = threadIdx.x & 31, w = threadIdx.x >> 5;
  if (lane == 0) red[w] = m;
  __syncthreads();
  if (w == 0) {
    float r = (lane < 8) ? red[lane] : -3.4e38f;
    r = warpMax(r);
    if (lane == 0) bmax[blockIdx.x] = r;
  }
}

__global__ __launch_bounds__(256) void k_redmax(const float* __restrict__ bmax,
                                                float* __restrict__ out) {
  __shared__ float red[8];
  float m = -3.4e38f;
  for (int i = threadIdx.x; i < 2048; i += 256) m = fmaxf(m, bmax[i]);
  m = warpMax(m);
  int lane = threadIdx.x & 31, w = threadIdx.x >> 5;
  if (lane == 0) red[w] = m;
  __syncthreads();
  if (w == 0) {
    float r = (lane < 8) ? red[lane] : -3.4e38f;
    r = warpMax(r);
    if (lane == 0) out[0] = r;
  }
}

// ---------------- K = exp(-(M/maxM)/reg), M recomputed on the fly ----------------
__global__ void k_expK(const float* __restrict__ G, const float* __restrict__ P,
                       const float* __restrict__ sqa, const float* __restrict__ sqb,
                       const float* __restrict__ maxM, float* __restrict__ Km) {
  int idx = blockIdx.x * blockDim.x + threadIdx.x;
  int i = idx >> 11, j = idx & (N_S - 1);
  float Mv = sqa[i] + sqb[j] + G[idx] - REG_ * logf(P[idx] + EPS_);
  Km[idx] = expf(-Mv * INV_REG / maxM[0]);
}

// ---------------- Sinkhorn matvecs (K L2-resident) ----------------
// v[j] = q / sum_i K[i,j]*u[i]   -- thread per column, coalesced across lanes
__global__ void k_colmv(const float* __restrict__ Km, const float* __restrict__ u,
                        float* __restrict__ v) {
  int j = blockIdx.x * blockDim.x + threadIdx.x;
  float s = 0.f;
  for (int i = 0; i < N_S; ++i) s += Km[(size_t)i * N_S + j] * u[i];
  v[j] = PQ / s;
}

// u[i] = p / sum_j K[i,j]*v[j]   -- wave per row, float4 loads
__global__ __launch_bounds__(256) void k_rowmv(const float* __restrict__ Km,
                                               const float* __restrict__ v,
                                               float* __restrict__ u) {
  int wave = (blockIdx.x * 256 + threadIdx.x) >> 5;
  int lane = threadIdx.x & 31;
  const float* row = Km + (size_t)wave * N_S;
  float s = 0.f;
  for (int j = lane * 4; j < N_S; j += 128) {
    float4 kv = *(const float4*)(row + j);
    float4 vv = *(const float4*)(v + j);
    s += kv.x * vv.x + kv.y * vv.y + kv.z * vv.z + kv.w * vv.w;
  }
  s = warpSum(s);
  if (lane == 0) u[wave] = PQ / s;
}

__global__ void k_updP(const float* __restrict__ Km, const float* __restrict__ u,
                       const float* __restrict__ v, float* __restrict__ P) {
  int idx = blockIdx.x * blockDim.x + threadIdx.x;
  int i = idx >> 11, j = idx & (N_S - 1);
  P[idx] = u[i] * Km[idx] * v[j];
}

// ---------------- column sums for the collapsed divergence means ----------------
__global__ void k_csM(const float* __restrict__ G, const float* __restrict__ P,
                      const float* __restrict__ sqa, const float* __restrict__ sqb,
                      const float* __restrict__ maxM, float* __restrict__ csM) {
  int j = blockIdx.x * blockDim.x + threadIdx.x;
  float sb = sqb[j], s = 0.f;
  for (int i = 0; i < N_S; ++i) {
    size_t idx = (size_t)i * N_S + j;
    s += sqa[i] + sb + G[idx] - REG_ * logf(P[idx] + EPS_);
  }
  csM[j] = s / maxM[0];
}

__global__ void k_csTL(const float* __restrict__ P, float* __restrict__ csT,
                       float* __restrict__ csL) {
  int j = blockIdx.x * blockDim.x + threadIdx.x;
  float st = 0.f, sl = 0.f;
  for (int i = 0; i < N_S; ++i) {
    float p = P[(size_t)i * N_S + j];
    st += p;
    sl += logf(p + EPS_) - 1.0f;
  }
  csT[j] = st; csL[j] = sl;
}

// mean(T@M.T + reg*T@(logT-1).T) = (csT.csM + reg*csT.csL)/n^2
__global__ __launch_bounds__(256) void k_final(const float* __restrict__ cs,
                                               float* __restrict__ out) {
  __shared__ float red[8];
  float tm[3] = {0.f, 0.f, 0.f}, tl[3] = {0.f, 0.f, 0.f};
  for (int j = threadIdx.x; j < N_S; j += 256) {
#pragma unroll
    for (int m = 0; m < 3; ++m) {
      const float* b = cs + m * 3 * N_S;
      float t = b[j];
      tm[m] += t * b[N_S + j];
      tl[m] += t * b[2 * N_S + j];
    }
  }
  float term[3];
#pragma unroll
  for (int m = 0; m < 3; ++m) {
    float a  = blockSum256(tm[m], red);
    float b2 = blockSum256(tl[m], red);
    term[m] = a + REG_ * b2;
  }
  if (threadIdx.x == 0) {
    const float inv = 1.0f / ((float)N_S * (float)N_S);
    out[0] = term[1] * inv - 0.5f * (term[2] * inv + term[0] * inv);
  }
}

// ---------------- host orchestration ----------------
extern "C" void kernel_launch(void* const* d_in, const int* in_sizes, int n_in,
                              void* d_out, int out_size, void* d_ws, size_t ws_size,
                              hipStream_t stream) {
  (void)in_sizes; (void)n_in; (void)out_size; (void)ws_size;
  const float* x = (const float*)d_in[0];
  const float* y = (const float*)d_in[1];
  float* out = (float*)d_out;
  char* ws = (char*)d_ws;

  const size_t NN = (size_t)N_S * N_S;
  float*    G   = (float*)(ws);
  float*    P   = (float*)(ws + NN * 4);
  float*    Km  = (float*)(ws + 2 * NN * 4);
  _Float16* xh  = (_Float16*)(ws + 3 * NN * 4);
  _Float16* yh  = xh + (size_t)N_S * D_F;
  char* S = ws + 3 * NN * 4 + 2 * (size_t)N_S * D_F * 2;
  float* sqx  = (float*)(S);
  float* sqy  = (float*)(S + 8192);
  float* u    = (float*)(S + 16384);
  float* vv   = (float*)(S + 24576);
  float* bmax = (float*)(S + 32768);
  float* maxM = (float*)(S + 40960);
  float* cs   = (float*)(S + 49152);   // 3 mats x {csT, csM, csL} x 2048 floats

  const int nE = N_S * D_F;
  k_cvt<<<(nE + 255) / 256, 256, 0, stream>>>(x, xh, nE);
  k_cvt<<<(nE + 255) / 256, 256, 0, stream>>>(y, yh, nE);
  k_rowsq<<<N_S / 8, 256, 0, stream>>>(x, sqx);
  k_rowsq<<<N_S / 8, 256, 0, stream>>>(y, sqy);

  const _Float16* Am[3] = {xh, xh, yh};
  const _Float16* Bm[3] = {xh, yh, yh};
  const float*    sa[3] = {sqx, sqx, sqy};
  const float*    sb[3] = {sqx, sqy, sqy};

  dim3 gemmGrid(N_S / 128, N_S / 128);
  for (int m = 0; m < 3; ++m) {
    k_gemm<<<gemmGrid, 256, 0, stream>>>(Am[m], Bm[m], G);
    k_fill<<<(int)(NN / 256), 256, 0, stream>>>(P, 1.0f / (float)NN, (int)NN);
    for (int it = 0; it < PROX; ++it) {
      k_partmax<<<2048, 256, 0, stream>>>(G, P, sa[m], sb[m], bmax);
      k_redmax<<<1, 256, 0, stream>>>(bmax, maxM);
      if (it == PROX - 1)   // colsum of final (normalized) M, before P is overwritten
        k_csM<<<N_S / 256, 256, 0, stream>>>(G, P, sa[m], sb[m], maxM,
                                             cs + m * 3 * N_S + N_S);
      k_expK<<<(int)(NN / 256), 256, 0, stream>>>(G, P, sa[m], sb[m], maxM, Km);
      k_fill<<<N_S / 256, 256, 0, stream>>>(u, PQ, N_S);
      for (int s2 = 0; s2 < SINK; ++s2) {
        k_colmv<<<N_S / 256, 256, 0, stream>>>(Km, u, vv);
        k_rowmv<<<N_S / 8, 256, 0, stream>>>(Km, vv, u);
      }
      k_colmv<<<N_S / 256, 256, 0, stream>>>(Km, u, vv);   // final v
      k_updP<<<(int)(NN / 256), 256, 0, stream>>>(Km, u, vv, P);
    }
    k_csTL<<<N_S / 256, 256, 0, stream>>>(P, cs + m * 3 * N_S, cs + m * 3 * N_S + 2 * N_S);
  }
  k_final<<<1, 256, 0, stream>>>(cs, out);
}
